// PositionwiseFeedForward_37409165148609
// MI455X (gfx1250) — matrix operand order repaired and checked by reading, not compiled, not run on hardware
//
#include <hip/hip_runtime.h>

// ---------------------------------------------------------------------------
// Problem constants (from setup_inputs): B=8, S=2048, D=1024, F=2048, 5 types
// ---------------------------------------------------------------------------
#define TOK   16384   // B*S
#define DDIM  1024
#define FDIM  2048
#define NTYPES 5

typedef __bf16 bf16;
typedef __attribute__((ext_vector_type(16))) __bf16 v16bf;
typedef __attribute__((ext_vector_type(8)))  __bf16 bf16x8;   // 16 bytes
typedef __attribute__((ext_vector_type(8)))  float   v8f;
typedef __attribute__((ext_vector_type(4)))  float   f32x4;

struct alignas(8) bf16x4s { bf16 a, b, c, d; };

// ---------------------------------------------------------------------------
// CDNA5 async global->LDS copy (ASYNCcnt-tracked DMA, no VGPR bounce).
// GV addressing: 64-bit vaddr, LDS byte address in VDST VGPR.
// ---------------------------------------------------------------------------
__device__ __forceinline__ void async_ld16(unsigned lds_addr, const void* gaddr)
{
    asm volatile("global_load_async_to_lds_b128 %0, %1, off"
                 :: "v"(lds_addr), "v"((unsigned long long)(uintptr_t)gaddr)
                 : "memory");
}
__device__ __forceinline__ void wait_async_le9()
{
    asm volatile("s_wait_asynccnt 0x9" ::: "memory");
}
__device__ __forceinline__ void wait_async_0()
{
    asm volatile("s_wait_asynccnt 0x0" ::: "memory");
}

// ---------------------------------------------------------------------------
// Kernel 1: LayerNorm (f32) -> bf16 rows.  One 256-thread block per row.
// ---------------------------------------------------------------------------
__global__ __launch_bounds__(256)
void pwff_ln_kernel(const float* __restrict__ x,
                    const float* __restrict__ gamma,
                    const float* __restrict__ beta,
                    bf16* __restrict__ h)
{
    __shared__ float red[256];
    const int row = blockIdx.x;
    const int tid = threadIdx.x;
    const float* xr = x + (size_t)row * DDIM;

    f32x4 v = *(const f32x4*)(xr + tid * 4);
    red[tid] = v.x + v.y + v.z + v.w;
    __syncthreads();
    for (int o = 128; o > 0; o >>= 1) {
        if (tid < o) red[tid] += red[tid + o];
        __syncthreads();
    }
    const float mean = red[0] * (1.0f / DDIM);
    __syncthreads();

    const float d0 = v.x - mean, d1 = v.y - mean, d2 = v.z - mean, d3 = v.w - mean;
    red[tid] = d0 * d0 + d1 * d1 + d2 * d2 + d3 * d3;
    __syncthreads();
    for (int o = 128; o > 0; o >>= 1) {
        if (tid < o) red[tid] += red[tid + o];
        __syncthreads();
    }
    const float rstd = rsqrtf(red[0] * (1.0f / DDIM) + 1e-6f);

    const f32x4 g = *(const f32x4*)(gamma + tid * 4);
    const f32x4 b = *(const f32x4*)(beta  + tid * 4);
    bf16x4s o;
    o.a = (bf16)(d0 * rstd * g.x + b.x);
    o.b = (bf16)(d1 * rstd * g.y + b.y);
    o.c = (bf16)(d2 * rstd * g.z + b.z);
    o.d = (bf16)(d3 * rstd * g.w + b.w);
    *(bf16x4s*)(h + (size_t)row * DDIM + tid * 4) = o;
}

// ---------------------------------------------------------------------------
// Kernel 2: tiled transpose + f32->bf16.  W:[K][N] row-major -> WT:[N][K].
// ---------------------------------------------------------------------------
__global__ __launch_bounds__(256)
void pwff_transpose_kernel(const float* __restrict__ W, bf16* __restrict__ WT,
                           int K, int N)
{
    __shared__ float tile[32][33];
    const int nt = blockIdx.x, kt = blockIdx.y;
    const int tx = threadIdx.x & 31;
    const int ty = threadIdx.x >> 5;          // 0..7
#pragma unroll
    for (int p = 0; p < 4; ++p) {
        int k = kt * 32 + ty + p * 8;
        tile[ty + p * 8][tx] = W[(size_t)k * N + nt * 32 + tx];
    }
    __syncthreads();
#pragma unroll
    for (int p = 0; p < 4; ++p) {
        int n = nt * 32 + ty + p * 8;
        WT[(size_t)n * K + kt * 32 + tx] = (bf16)tile[tx][ty + p * 8];
    }
}

// ---------------------------------------------------------------------------
// Kernel 3a/3b: bucket tokens by code type.
// ---------------------------------------------------------------------------
__global__ void pwff_zero_cnt_kernel(int* cnt)
{
    if (threadIdx.x < NTYPES) cnt[threadIdx.x] = 0;
}

__global__ __launch_bounds__(256)
void pwff_bucket_kernel(const int* __restrict__ code, int* __restrict__ cnt,
                        int* __restrict__ idxb)
{
    const int i = blockIdx.x * 256 + threadIdx.x;
    if (i < TOK) {
        const int t = code[i];
        const int slot = atomicAdd(&cnt[t], 1);
        idxb[t * TOK + slot] = i;
    }
}

// ---------------------------------------------------------------------------
// Kernel 4: bucketed GEMM, bf16 WMMA, f32 accumulate.
//   Block = 256 threads = 8 waves (2 M-waves x 4 N-waves).
//   Block tile: 32 tokens x 256 cols; K chunks of 64 double-buffered in LDS
//   and filled with global_load_async_to_lds_b128 (ASYNCcnt pipelined).
// ---------------------------------------------------------------------------
template<int K_DIM, int N_TYPE, bool STAGE2>
__global__ __launch_bounds__(256)
void pwff_gemm_kernel(const bf16*  __restrict__ A,      // [TOK][K_DIM]
                      const bf16*  __restrict__ BT,     // [5*N_TYPE][K_DIM]
                      const float* __restrict__ bias,   // [5*N_TYPE]
                      const float* __restrict__ resid,  // [TOK][N_TYPE] or null
                      float*       __restrict__ outF,
                      bf16*        __restrict__ outB,
                      const int*   __restrict__ cnt,
                      const int*   __restrict__ idxb)
{
    __shared__ alignas(32) bf16 sA[2][32][64];    //  8 KB
    __shared__ alignas(32) bf16 sB[2][256][64];   // 64 KB
    __shared__ int s_tok[32];

    const int t = blockIdx.z;
    const int n_count = cnt[t];
    const int mt = blockIdx.y;
    if (mt * 32 >= n_count) return;

    const int tid  = threadIdx.x;
    const int lane = tid & 31;
    const int wid  = tid >> 5;

    if (tid < 32) {
        int row = mt * 32 + tid;
        int r = row < n_count ? row : (n_count - 1);   // clamp pad rows
        s_tok[tid] = idxb[t * TOK + r];
    }
    __syncthreads();

    const int nBase = blockIdx.x * 256;                       // type-local N base
    const size_t colBase = (size_t)t * N_TYPE + nBase;        // row index into BT

    // ---- staging assignments ----
    const int am   = tid >> 3;          // A: row 0..31
    const int aseg = tid & 7;           // A: 16B segment
    const bf16* aRow = A + (size_t)s_tok[am] * K_DIM + aseg * 8;

    const int bn0  = tid >> 3;          // B: rows bn0 + 32*r
    const int bseg = tid & 7;
    const bf16* bRow[8];
#pragma unroll
    for (int r = 0; r < 8; ++r)
        bRow[r] = BT + (colBase + bn0 + 32 * r) * (size_t)K_DIM + bseg * 8;

    // LDS byte addresses for this thread's async destinations
    const unsigned ldsA = (unsigned)(uintptr_t)&sA[0][0][0] +
                          (unsigned)(am * 64 + aseg * 8) * 2u;
    const unsigned ldsB = (unsigned)(uintptr_t)&sB[0][0][0] +
                          (unsigned)(bn0 * 64 + bseg * 8) * 2u;
    const unsigned sA_STRIDE = 32 * 64 * 2;    // per-buffer bytes
    const unsigned sB_STRIDE = 256 * 64 * 2;

    v8f acc[4] = {};

    const int halfSel = lane >> 4;              // 0: lanes 0-15, 1: lanes 16-31
    const int lm      = lane & 15;
    const int aRowL   = (wid >> 2) * 16 + lm;   // wave M row in block tile
    const int nW      = (wid & 3) * 64;         // wave N base in block tile

    const int NCHUNK = K_DIM / 64;

    // ---- prologue: async-fill buffer 0 ----
    {
        async_ld16(ldsA, aRow);
#pragma unroll
        for (int r = 0; r < 8; ++r)
            async_ld16(ldsB + (unsigned)(r * 32 * 64 * 2), bRow[r]);
    }

    for (int i = 0; i < NCHUNK; ++i) {
        const int buf = i & 1;
        // issue next chunk into the other buffer, then drain this chunk's 9 ops
        if (i + 1 < NCHUNK) {
            const int nb = (i + 1) & 1;
            const int kk = (i + 1) * 64;
            async_ld16(ldsA + nb * sA_STRIDE, aRow + kk);
#pragma unroll
            for (int r = 0; r < 8; ++r)
                async_ld16(ldsB + nb * sB_STRIDE + (unsigned)(r * 32 * 64 * 2),
                           bRow[r] + kk);
            wait_async_le9();
        } else {
            wait_async_0();
        }
        __syncthreads();

        // ---- preload all fragments for this K=64 chunk, then 8 WMMAs ----
        union FA { v16bf v; bf16x8 h[2]; } fa0, fa1;
        fa0.h[0] = *(const bf16x8*)&sA[buf][aRowL][ 0 + halfSel * 16];
        fa0.h[1] = *(const bf16x8*)&sA[buf][aRowL][ 8 + halfSel * 16];
        fa1.h[0] = *(const bf16x8*)&sA[buf][aRowL][32 + halfSel * 16];
        fa1.h[1] = *(const bf16x8*)&sA[buf][aRowL][40 + halfSel * 16];

        v16bf fb0[4], fb1[4];
#pragma unroll
        for (int j = 0; j < 4; ++j) {
            fb0[j] = *(const v16bf*)&sB[buf][nW + j * 16 + lm][ 0 + halfSel * 16];
            fb1[j] = *(const v16bf*)&sB[buf][nW + j * 16 + lm][32 + halfSel * 16];
        }
#pragma unroll
        for (int j = 0; j < 4; ++j)
            acc[j] = __builtin_amdgcn_wmma_f32_16x16x32_bf16(
                false, fa0.v, false, fb0[j], (short)0, acc[j], false, false);
#pragma unroll
        for (int j = 0; j < 4; ++j)
            acc[j] = __builtin_amdgcn_wmma_f32_16x16x32_bf16(
                false, fa1.v, false, fb1[j], (short)0, acc[j], false, false);
        __syncthreads();
    }

    // ---- epilogue: C/D layout — lane holds column lm; VGPR v -> row v (+8
    // for lanes 16-31) of the wave's 16-row tile. ----
    const int mBase = (wid >> 2) * 16 + halfSel * 8;
#pragma unroll
    for (int j = 0; j < 4; ++j) {
        const int nLocal = nBase + nW + j * 16 + lm;          // within N_TYPE
        const float bv = bias[(size_t)t * N_TYPE + nLocal];
#pragma unroll
        for (int v = 0; v < 8; ++v) {
            const int m = mBase + v;
            if (mt * 32 + m < n_count) {
                const int tok = s_tok[m];
                float val = acc[j][v] + bv;
                if (STAGE2) {
                    val += resid[(size_t)tok * N_TYPE + nLocal];
                    outF[(size_t)tok * N_TYPE + nLocal] = val;
                } else {
                    val = val > 0.0f ? val : 0.0f;
                    outB[(size_t)tok * N_TYPE + nLocal] = (bf16)val;
                }
            }
        }
    }
}

// ---------------------------------------------------------------------------
// Launcher
// ---------------------------------------------------------------------------
extern "C" void kernel_launch(void* const* d_in, const int* in_sizes, int n_in,
                              void* d_out, int out_size, void* d_ws, size_t ws_size,
                              hipStream_t stream)
{
    const float* x     = (const float*)d_in[0];   // [TOK][D]
    const int*   code  = (const int*)  d_in[1];   // [TOK]
    const float* W1    = (const float*)d_in[2];   // [D][5F]
    const float* b1    = (const float*)d_in[3];   // [5F]
    const float* W2    = (const float*)d_in[4];   // [F][5D]
    const float* b2    = (const float*)d_in[5];   // [5D]
    const float* gamma = (const float*)d_in[6];   // [D]
    const float* beta  = (const float*)d_in[7];   // [D]
    float* out = (float*)d_out;                   // [TOK][D]

    // Workspace layout (bytes):
    char* ws = (char*)d_ws;
    const size_t H_B    = (size_t)TOK * DDIM * sizeof(bf16);            // 32 MB
    const size_t W1T_B  = (size_t)NTYPES * FDIM * DDIM * sizeof(bf16);  // 20 MB
    const size_t W2T_B  = (size_t)NTYPES * DDIM * FDIM * sizeof(bf16);  // 20 MB
    const size_t INT_B  = (size_t)TOK * FDIM * sizeof(bf16);            // 64 MB
    const size_t IDX_B  = (size_t)NTYPES * TOK * sizeof(int);           // 320 KB

    bf16* hbf     = (bf16*)(ws);
    bf16* w1t     = (bf16*)(ws + H_B);
    bf16* w2t     = (bf16*)(ws + H_B + W1T_B);
    bf16* interbf = (bf16*)(ws + H_B + W1T_B + W2T_B);
    int*  idxb    = (int*) (ws + H_B + W1T_B + W2T_B + INT_B);
    int*  cnt     = (int*) (ws + H_B + W1T_B + W2T_B + INT_B + IDX_B);

    // 1) LayerNorm -> bf16
    pwff_ln_kernel<<<TOK, 256, 0, stream>>>(x, gamma, beta, hbf);

    // 2) Weight transpose+convert (weights stay resident in L2 afterwards)
    pwff_transpose_kernel<<<dim3((NTYPES * FDIM) / 32, DDIM / 32), 256, 0, stream>>>(
        W1, w1t, DDIM, NTYPES * FDIM);
    pwff_transpose_kernel<<<dim3((NTYPES * DDIM) / 32, FDIM / 32), 256, 0, stream>>>(
        W2, w2t, FDIM, NTYPES * DDIM);

    // 3) Bucket tokens by type
    pwff_zero_cnt_kernel<<<1, 32, 0, stream>>>(cnt);
    pwff_bucket_kernel<<<TOK / 256, 256, 0, stream>>>(code, cnt, idxb);

    // 4) GEMM1: inter = relu(h @ W1_t + b1)  (bf16 out)
    pwff_gemm_kernel<DDIM, FDIM, false>
        <<<dim3(FDIM / 256, TOK / 32, NTYPES), 256, 0, stream>>>(
            hbf, w1t, b1, nullptr, nullptr, interbf, cnt, idxb);

    // 5) GEMM2: out = inter @ W2_t + b2 + x  (f32 out)
    pwff_gemm_kernel<FDIM, DDIM, true>
        <<<dim3(DDIM / 256, TOK / 32, NTYPES), 256, 0, stream>>>(
            interbf, w2t, b2, x, out, nullptr, cnt, idxb);
}
